// BayesDGN_Conv_25675314495759
// MI455X (gfx1250) — compile-verified, matrix operand order checked by reading, not run on hardware
//
#include <hip/hip_runtime.h>
#include <hip/hip_bf16.h>

#define NN   50000
#define NE   600000
#define OBS  256
#define EHID 512
#define HD   128
#define NH   8
#define HDIM 16
#define OUTW 384

typedef __attribute__((ext_vector_type(16))) _Float16 v16h;
typedef __attribute__((ext_vector_type(8)))  _Float16 v8h;
typedef __attribute__((ext_vector_type(8)))  float    v8f;
typedef __attribute__((ext_vector_type(4)))  float    v4f;

// ---------------- WMMA fragment helpers (CDNA5 16x16x32 f16, wave32) ----------
// A layout (ISA 7.12.2): lane&15 = M row; lanes 0-15 hold K {0..7,16..23},
// lanes 16-31 hold K {8..15,24..31}. B (from f16 [N][K] transposed weights):
// lane&15 = N column, same K mapping. C/D: VGPR i -> row (i + 8*(lane>>4)),
// lane&15 = N column.

__device__ inline v16h a_frag_f32(const float* __restrict__ A, int lda, int m0, int k0) {
  const int lane = threadIdx.x & 31;
  const float* p = A + (long)(m0 + (lane & 15)) * lda + (k0 + ((lane & 16) ? 8 : 0));
  v4f x0 = *(const v4f*)(p);
  v4f x1 = *(const v4f*)(p + 4);
  v4f x2 = *(const v4f*)(p + 16);
  v4f x3 = *(const v4f*)(p + 20);
  v16h a;
#pragma unroll
  for (int i = 0; i < 4; ++i) {
    a[i]      = (_Float16)x0[i];
    a[i + 4]  = (_Float16)x1[i];
    a[i + 8]  = (_Float16)x2[i];
    a[i + 12] = (_Float16)x3[i];
  }
  return a;
}

__device__ inline v16h frag_f16(const _Float16* __restrict__ B, int ldk, int n0, int k0) {
  const int lane = threadIdx.x & 31;
  const _Float16* p = B + (long)(n0 + (lane & 15)) * ldk + (k0 + ((lane & 16) ? 8 : 0));
  v8h lo = *(const v8h*)(p);        // K +0..7   (16B)
  v8h hi = *(const v8h*)(p + 16);   // K +16..23 (16B)
  v16h b;
#pragma unroll
  for (int i = 0; i < 8; ++i) { b[i] = lo[i]; b[i + 8] = hi[i]; }
  return b;
}

// ---------------- weight convert + transpose (f32 [K][N] -> f16 [N][K]) ------
__global__ void convt_kernel(const float* __restrict__ W, _Float16* __restrict__ Wt,
                             int K, int N) {
  int idx = blockIdx.x * blockDim.x + threadIdx.x;
  if (idx >= K * N) return;
  int k = idx / N, n = idx - k * N;
  Wt[(long)n * K + k] = (_Float16)W[idx];
}

// ---------------- fused 2-layer MLP encoder ----------------------------------
// block = 128 threads = 4 waves, one 16-row tile of nodes.
// phase1: hid[16,512] = relu(obs@W1+b1) -> f16 LDS (wave w covers cols w*128..+127)
// phase2: z1[16,128]  = relu(hid@W2+b2) -> out cols 0..127 (wave w covers 32 cols)
__global__ __launch_bounds__(128) void enc_kernel(
    const float* __restrict__ obs, const _Float16* __restrict__ W1t,
    const float* __restrict__ b1,  const _Float16* __restrict__ W2t,
    const float* __restrict__ b2,  float* __restrict__ out) {
  __shared__ _Float16 hid[16 * EHID];
  const int wave = threadIdx.x >> 5;
  const int lane = threadIdx.x & 31;
  const int m0 = blockIdx.x * 16;
  const int n = lane & 15;
  const int mb = (lane & 16) ? 8 : 0;

  v8f acc[8] = {};
  for (int k0 = 0; k0 < OBS; k0 += 32) {
    v16h a = a_frag_f32(obs, OBS, m0, k0);
#pragma unroll
    for (int t = 0; t < 8; ++t) {
      v16h b = frag_f16(W1t, OBS, (wave * 8 + t) * 16, k0);
      acc[t] = __builtin_amdgcn_wmma_f32_16x16x32_f16(false, a, false, b,
                                                      (short)0, acc[t], false, false);
    }
  }
#pragma unroll
  for (int t = 0; t < 8; ++t) {
    int col = (wave * 8 + t) * 16 + n;
    float bv = b1[col];
#pragma unroll
    for (int i = 0; i < 8; ++i) {
      float v = acc[t][i] + bv;
      hid[(mb + i) * EHID + col] = (_Float16)(v > 0.f ? v : 0.f);
    }
  }
  __syncthreads();

  v8f acc2[2] = {};
  for (int k0 = 0; k0 < EHID; k0 += 32) {
    v16h a = frag_f16(hid, EHID, 0, k0);   // A from LDS, same lane mapping
#pragma unroll
    for (int t = 0; t < 2; ++t) {
      v16h b = frag_f16(W2t, EHID, (wave * 2 + t) * 16, k0);
      acc2[t] = __builtin_amdgcn_wmma_f32_16x16x32_f16(false, a, false, b,
                                                       (short)0, acc2[t], false, false);
    }
  }
#pragma unroll
  for (int t = 0; t < 2; ++t) {
    int col = (wave * 2 + t) * 16 + n;
    float bv = b2[col];
#pragma unroll
    for (int i = 0; i < 8; ++i) {
      float v = acc2[t][i] + bv;
      out[(long)(m0 + mb + i) * OUTW + col] = v > 0.f ? v : 0.f;
    }
  }
}

// ---------------- GAT feature GEMM: feat = H @ Wg  (H strided in d_out) ------
__global__ __launch_bounds__(128) void feat_kernel(
    const float* __restrict__ H, const _Float16* __restrict__ Wt,
    float* __restrict__ feat) {
  const int wave = threadIdx.x >> 5;
  const int lane = threadIdx.x & 31;
  const int m0 = blockIdx.x * 16;
  const int n = lane & 15;
  const int mb = (lane & 16) ? 8 : 0;
  v8f acc[2] = {};
  for (int k0 = 0; k0 < HD; k0 += 32) {
    v16h a = a_frag_f32(H, OUTW, m0, k0);
#pragma unroll
    for (int t = 0; t < 2; ++t) {
      v16h b = frag_f16(Wt, HD, (wave * 2 + t) * 16, k0);
      acc[t] = __builtin_amdgcn_wmma_f32_16x16x32_f16(false, a, false, b,
                                                      (short)0, acc[t], false, false);
    }
  }
#pragma unroll
  for (int t = 0; t < 2; ++t) {
    int col = (wave * 2 + t) * 16 + n;
#pragma unroll
    for (int i = 0; i < 8; ++i)
      feat[(long)(m0 + mb + i) * HD + col] = acc[t][i];
  }
}

// ---------------- attention logits per (node, head) --------------------------
__global__ void eler_kernel(const float* __restrict__ feat, const float* __restrict__ al,
                            const float* __restrict__ ar, float* __restrict__ el,
                            float* __restrict__ er) {
  int idx = blockIdx.x * blockDim.x + threadIdx.x;
  if (idx >= NN * NH) return;
  int h = idx & 7, nd = idx >> 3;
  const float* f  = feat + (long)nd * HD + h * HDIM;
  const float* pl = al + h * HDIM;
  const float* pr = ar + h * HDIM;
  float sl = 0.f, sr = 0.f;
#pragma unroll
  for (int d = 0; d < HDIM; ++d) { float fv = f[d]; sl += fv * pl[d]; sr += fv * pr[d]; }
  el[idx] = sl; er[idx] = sr;
}

// ordered-uint encoding so unsigned atomicMax == float max (monotonic map)
__device__ inline unsigned ord_enc(float f) {
  unsigned u = __float_as_uint(f);
  return (u & 0x80000000u) ? ~u : (u | 0x80000000u);
}
__device__ inline float ord_dec(unsigned u) {
  return (u & 0x80000000u) ? __uint_as_float(u ^ 0x80000000u) : __uint_as_float(~u);
}
__device__ inline float lrelu(float x) { return x > 0.f ? x : 0.2f * x; }

__global__ void init_kernel(unsigned* __restrict__ m, float* __restrict__ z,
                            float* __restrict__ acc) {
  int idx = blockIdx.x * blockDim.x + threadIdx.x;
  if (idx >= NN * HD) return;
  acc[idx] = 0.f;
  if (idx < NN * NH) { m[idx] = 0x007FFFFFu; z[idx] = 0.f; }  // enc(-inf)
}

__global__ void edge_max_kernel(const int* __restrict__ src, const int* __restrict__ dst,
                                const float* __restrict__ el, const float* __restrict__ er,
                                unsigned* __restrict__ m) {
  int e = blockIdx.x * blockDim.x + threadIdx.x;
  if (e >= NE) return;
  long s = src[e], d = dst[e];
  v4f el0 = *(const v4f*)(el + s * 8), el1 = *(const v4f*)(el + s * 8 + 4);
  v4f er0 = *(const v4f*)(er + d * 8), er1 = *(const v4f*)(er + d * 8 + 4);
#pragma unroll
  for (int h = 0; h < 4; ++h) {
    atomicMax(&m[d * 8 + h],     ord_enc(lrelu(el0[h] + er0[h])));
    atomicMax(&m[d * 8 + 4 + h], ord_enc(lrelu(el1[h] + er1[h])));
  }
}

__global__ void edge_sum_kernel(const int* __restrict__ src, const int* __restrict__ dst,
                                const float* __restrict__ el, const float* __restrict__ er,
                                const unsigned* __restrict__ m, float* __restrict__ z) {
  int e = blockIdx.x * blockDim.x + threadIdx.x;
  if (e >= NE) return;
  long s = src[e], d = dst[e];
  v4f el0 = *(const v4f*)(el + s * 8), el1 = *(const v4f*)(el + s * 8 + 4);
  v4f er0 = *(const v4f*)(er + d * 8), er1 = *(const v4f*)(er + d * 8 + 4);
#pragma unroll
  for (int h = 0; h < 4; ++h) {
    float e0 = lrelu(el0[h] + er0[h]);
    float e1 = lrelu(el1[h] + er1[h]);
    unsafeAtomicAdd(&z[d * 8 + h],     __expf(e0 - ord_dec(m[d * 8 + h])));
    unsafeAtomicAdd(&z[d * 8 + 4 + h], __expf(e1 - ord_dec(m[d * 8 + 4 + h])));
  }
}

// one thread per (edge, head): scatter-add alpha-weighted 16-dim head feature
__global__ void edge_msg_kernel(const int* __restrict__ src, const int* __restrict__ dst,
                                const float* __restrict__ el, const float* __restrict__ er,
                                const unsigned* __restrict__ m, const float* __restrict__ z,
                                const float* __restrict__ feat, float* __restrict__ acc) {
  int idx = blockIdx.x * blockDim.x + threadIdx.x;
  if (idx >= NE * NH) return;
  int h = idx & 7, e = idx >> 3;
  long s = src[e], d = dst[e];
  float ev = lrelu(el[s * 8 + h] + er[d * 8 + h]);
  float ex = __expf(ev - ord_dec(m[d * 8 + h]));
  float alpha = ex / (z[d * 8 + h] + 1e-9f);
  const float* fs = feat + s * HD + h * HDIM;
  float* ad = acc + d * HD + h * HDIM;
#pragma unroll
  for (int dd = 0; dd < HDIM; ++dd) unsafeAtomicAdd(&ad[dd], fs[dd] * alpha);
}

__global__ void finalize_kernel(const float* __restrict__ acc, float* __restrict__ out,
                                int coloff) {
  int idx = blockIdx.x * blockDim.x + threadIdx.x;
  if (idx >= NN * HD) return;
  int nd = idx >> 7, c = idx & 127;
  float v = acc[idx];
  out[(long)nd * OUTW + coloff + c] = v > 0.f ? v : 0.f;
}

// ---------------------------------------------------------------------------
extern "C" void kernel_launch(void* const* d_in, const int* in_sizes, int n_in,
                              void* d_out, int out_size, void* d_ws, size_t ws_size,
                              hipStream_t stream) {
  const float* obs = (const float*)d_in[0];
  const int*   src = (const int*)d_in[1];
  const int*   dst = (const int*)d_in[2];
  const float* W1  = (const float*)d_in[3];
  const float* b1  = (const float*)d_in[4];
  const float* W2  = (const float*)d_in[5];
  const float* b2  = (const float*)d_in[6];
  const float* Wg1 = (const float*)d_in[7];
  const float* al1 = (const float*)d_in[8];
  const float* ar1 = (const float*)d_in[9];
  const float* Wg2 = (const float*)d_in[10];
  const float* al2 = (const float*)d_in[11];
  const float* ar2 = (const float*)d_in[12];
  float* out = (float*)d_out;

  char* ws = (char*)d_ws;
  _Float16* W1t  = (_Float16*)(ws + 0);          // 512*256*2 = 262144
  _Float16* W2t  = (_Float16*)(ws + 262144);     // 128*512*2 = 131072
  _Float16* Wg1t = (_Float16*)(ws + 393216);     // 128*128*2 = 32768
  _Float16* Wg2t = (_Float16*)(ws + 425984);     // 32768
  float*    feat = (float*)(ws + 458752);        // 50000*128*4 = 25600000
  float*    el   = (float*)(ws + 26058752);      // 1600000
  float*    er   = (float*)(ws + 27658752);      // 1600000
  unsigned* mbuf = (unsigned*)(ws + 29258752);   // 1600000
  float*    zbuf = (float*)(ws + 30858752);      // 1600000
  float*    acc  = (float*)(ws + 32458752);      // 25600000  (total ~58 MB)

  const int TB = 256;
  convt_kernel<<<(OBS * EHID + TB - 1) / TB, TB, 0, stream>>>(W1, W1t, OBS, EHID);
  convt_kernel<<<(EHID * HD + TB - 1) / TB, TB, 0, stream>>>(W2, W2t, EHID, HD);
  convt_kernel<<<(HD * HD + TB - 1) / TB, TB, 0, stream>>>(Wg1, Wg1t, HD, HD);
  convt_kernel<<<(HD * HD + TB - 1) / TB, TB, 0, stream>>>(Wg2, Wg2t, HD, HD);

  enc_kernel<<<NN / 16, 128, 0, stream>>>(obs, W1t, b1, W2t, b2, out);

  for (int layer = 0; layer < 2; ++layer) {
    const _Float16* Wt = layer ? Wg2t : Wg1t;
    const float* al = layer ? al2 : al1;
    const float* ar = layer ? ar2 : ar1;
    const int cin  = layer ? HD : 0;       // read z2 or z1 from d_out
    const int cout = layer ? 2 * HD : HD;  // write z3 or z2 into d_out

    feat_kernel<<<NN / 16, 128, 0, stream>>>(out + cin, Wt, feat);
    eler_kernel<<<(NN * NH + TB - 1) / TB, TB, 0, stream>>>(feat, al, ar, el, er);
    init_kernel<<<(NN * HD + TB - 1) / TB, TB, 0, stream>>>(mbuf, zbuf, acc);
    edge_max_kernel<<<(NE + TB - 1) / TB, TB, 0, stream>>>(src, dst, el, er, mbuf);
    edge_sum_kernel<<<(NE + TB - 1) / TB, TB, 0, stream>>>(src, dst, el, er, mbuf, zbuf);
    edge_msg_kernel<<<(NE * NH + TB - 1) / TB, TB, 0, stream>>>(src, dst, el, er, mbuf,
                                                                zbuf, feat, acc);
    finalize_kernel<<<(NN * HD + TB - 1) / TB, TB, 0, stream>>>(acc, out, cout);
  }
}
